// MSC_Modified_17669495455797
// MI455X (gfx1250) — matrix operand order repaired and checked by reading, not compiled, not run on hardware
//
#include <hip/hip_runtime.h>
#include <hip/hip_bf16.h>

// ---------------------------------------------------------------------------
// MI455X (gfx1250, wave32) implementation of the multi-scale cross-attention
// block. All matmuls (q/kv/proj projections, Q@K^T, P@V) run on
// v_wmma_f32_16x16x32_f16. Attention score rows (16 x 1024 f32) stay in LDS;
// top-k thresholds are found with an exact radix-select on ordered float bits.
// V is produced pre-transposed ([B*C, N]) so P@V B-fragments load as b128.
// ---------------------------------------------------------------------------

typedef __attribute__((ext_vector_type(16))) _Float16 v16h;
typedef __attribute__((ext_vector_type(8)))  float    v8f;

#define BDIM   8
#define CDIM   512
#define HDIM   32
#define WDIM   32
#define NSEQ   1024          // H*W
#define NHEADS 8
#define HD     64            // head dim
#define MROWS  8192          // B * NSEQ

// ---------------- order-preserving float <-> uint mapping -------------------
__device__ __forceinline__ unsigned f2ord(float f) {
  unsigned u = __float_as_uint(f);
  return (u & 0x80000000u) ? ~u : (u | 0x80000000u);
}
__device__ __forceinline__ float ord2f(unsigned u) {
  unsigned b = (u & 0x80000000u) ? (u & 0x7fffffffu) : ~u;
  return __uint_as_float(b);
}

__device__ __forceinline__ float wave_max(float v) {
  #pragma unroll
  for (int off = 16; off; off >>= 1) v = fmaxf(v, __shfl_xor(v, off));
  return v;
}
__device__ __forceinline__ float wave_sum(float v) {
  #pragma unroll
  for (int off = 16; off; off >>= 1) v += __shfl_xor(v, off);
  return v;
}

// exact k-th largest of n floats in an LDS row, via 8-bit radix on ordered bits
__device__ float radix_kth_largest(const float* rowp, int n, int k,
                                   unsigned* hist /*256, per-wave*/, int lane) {
  unsigned prefix = 0u;
  int remaining = k;
  for (int shift = 24; shift >= 0; shift -= 8) {
    for (int i = lane; i < 256; i += 32) hist[i] = 0u;
    __threadfence_block();
    unsigned mask = (shift == 24) ? 0u : (0xFFFFFFFFu << (shift + 8));
    for (int j = lane; j < n; j += 32) {
      unsigned u = f2ord(rowp[j]);
      if ((u & mask) == prefix) atomicAdd(&hist[(u >> shift) & 255u], 1u);
    }
    __threadfence_block();
    unsigned sel = 0u; int rem2 = remaining;
    if (lane == 0) {
      int acc = 0;
      for (int v = 255; v >= 0; --v) {
        int c = (int)hist[v];
        if (acc + c >= remaining) { sel = (unsigned)v; rem2 = remaining - acc; break; }
        acc += c;
      }
    }
    sel = (unsigned)__shfl((int)sel, 0);
    rem2 = __shfl(rem2, 0);
    prefix |= sel << shift;
    remaining = rem2;
  }
  return ord2f(prefix);
}

// ---------------- small utility kernels -------------------------------------
__global__ void cvt_w_kernel(const float* __restrict__ qw,
                             const float* __restrict__ kvw,
                             const float* __restrict__ pw,
                             _Float16* __restrict__ qwh,
                             _Float16* __restrict__ kvwh,
                             _Float16* __restrict__ pwh) {
  int i = blockIdx.x * 256 + threadIdx.x;
  if (i < CDIM * CDIM) qwh[i] = (_Float16)qw[i];
  if (i < 2 * CDIM * CDIM) kvwh[i] = (_Float16)kvw[i];
  if (i < CDIM * CDIM) pwh[i] = (_Float16)pw[i];
}

__global__ void compute_k_kernel(const float* __restrict__ r1,
                                 const float* __restrict__ r2,
                                 int* __restrict__ kv) {
  if (threadIdx.x == 0) {
    float s1 = 1.0f / (1.0f + __expf(-r1[0]));
    float s2 = 1.0f / (1.0f + __expf(-r2[0]));
    int k1 = (int)((float)NSEQ * s1);
    int k2 = (int)((float)NSEQ * s2);
    k1 = k1 < 1 ? 1 : (k1 > NSEQ ? NSEQ : k1);
    k2 = k2 < 1 ? 1 : (k2 > NSEQ ? NSEQ : k2);
    kv[0] = k1; kv[1] = k2;
  }
}

__global__ void transpose_x_kernel(const float* __restrict__ x,
                                   _Float16* __restrict__ xs) {
  size_t i = (size_t)blockIdx.x * 256 + threadIdx.x;  // over B*C*N
  size_t n = i & (NSEQ - 1);
  size_t c = (i >> 10) & (CDIM - 1);
  size_t b = i >> 19;
  xs[((b << 10) + n) * CDIM + c] = (_Float16)x[i];
}

// ---------------- fused multiscale depthwise conv + LayerNorm ---------------
__global__ __launch_bounds__(512)
void conv_ln_kernel(const float* __restrict__ y,
                    const float* __restrict__ c1w, const float* __restrict__ c1b,
                    const float* __restrict__ c2w, const float* __restrict__ c2b,
                    const float* __restrict__ c3w, const float* __restrict__ c3b,
                    const float* __restrict__ lnw, const float* __restrict__ lnb,
                    _Float16* __restrict__ yn) {
  __shared__ float red[CDIM];
  int bn = blockIdx.x;
  int b = bn >> 10, nsp = bn & (NSEQ - 1);
  int hh = nsp >> 5, ww = nsp & 31;
  int c = threadIdx.x;
  const float* yp = y + ((size_t)b * CDIM + c) * NSEQ;   // 32x32 plane of chan c
  float acc = c1b[c] + c2b[c] + c3b[c];
  const float* w1 = c1w + c * 9;
  #pragma unroll
  for (int dy = -1; dy <= 1; ++dy)
    #pragma unroll
    for (int dx = -1; dx <= 1; ++dx) {
      int yy = hh + dy, xx = ww + dx;
      if (yy >= 0 && yy < HDIM && xx >= 0 && xx < WDIM)
        acc += yp[yy * WDIM + xx] * w1[(dy + 1) * 3 + (dx + 1)];
    }
  const float* w2 = c2w + c * 25;
  #pragma unroll
  for (int dy = -2; dy <= 2; ++dy)
    #pragma unroll
    for (int dx = -2; dx <= 2; ++dx) {
      int yy = hh + dy, xx = ww + dx;
      if (yy >= 0 && yy < HDIM && xx >= 0 && xx < WDIM)
        acc += yp[yy * WDIM + xx] * w2[(dy + 2) * 5 + (dx + 2)];
    }
  const float* w3 = c3w + c * 49;
  #pragma unroll
  for (int dy = -3; dy <= 3; ++dy)
    #pragma unroll
    for (int dx = -3; dx <= 3; ++dx) {
      int yy = hh + dy, xx = ww + dx;
      if (yy >= 0 && yy < HDIM && xx >= 0 && xx < WDIM)
        acc += yp[yy * WDIM + xx] * w3[(dy + 3) * 7 + (dx + 3)];
    }
  // LayerNorm over C
  red[c] = acc; __syncthreads();
  for (int s = CDIM / 2; s > 0; s >>= 1) { if (c < s) red[c] += red[c + s]; __syncthreads(); }
  float mu = red[0] * (1.0f / CDIM);
  __syncthreads();
  float d = acc - mu;
  red[c] = d * d; __syncthreads();
  for (int s = CDIM / 2; s > 0; s >>= 1) { if (c < s) red[c] += red[c + s]; __syncthreads(); }
  float var = red[0] * (1.0f / CDIM);
  float inv = rsqrtf(var + 1e-5f);
  float val = d * inv * lnw[c] + lnb[c];
  yn[((size_t)b * NSEQ + nsp) * CDIM + c] = (_Float16)val;
}

// ---------------- WMMA GEMM: out[M,512] = A[M,512] @ W[512,512]^T -----------
__global__ __launch_bounds__(128)
void gemm_wt_kernel(const _Float16* __restrict__ A,
                    const _Float16* __restrict__ W,    // [cols][512] row-major
                    _Float16* __restrict__ out, float scale) {
  __shared__ _Float16 sA[16][CDIM];                    // 16 KB A-tile
  const int row0 = blockIdx.x * 16;
  const int wave = threadIdx.x >> 5;
  const int lane = threadIdx.x & 31;
  for (int i = threadIdx.x; i < 16 * 64; i += 128) {
    int r = i >> 6, ch = i & 63;
    ((uint4*)&sA[r][0])[ch] = ((const uint4*)(A + (size_t)(row0 + r) * CDIM))[ch];
  }
  __syncthreads();
  const int n = lane & 15, g = lane >> 4, m = lane & 15;
  const int c0 = blockIdx.y * 64 + wave * 16;
  const _Float16* wr = W + (size_t)(c0 + n) * CDIM;    // column c0+n of W^T
  v8f acc = {};
  for (int k0 = 0; k0 < CDIM; k0 += 32) {
    __builtin_prefetch(wr + k0 + 128, 0, 3);           // near-cache prefetch
    v16h a, bw;
    #pragma unroll
    for (int v = 0; v < 8; ++v) {
      int ka = g * 16 + 2 * v;                         // A 16x32 layout
      a[2 * v]     = sA[m][k0 + ka];
      a[2 * v + 1] = sA[m][k0 + ka + 1];
      int kb = g * 16 + 2 * v;                         // B 32x16 layout
      bw[2 * v]     = wr[k0 + kb];
      bw[2 * v + 1] = wr[k0 + kb + 1];
    }
    acc = __builtin_amdgcn_wmma_f32_16x16x32_f16(false, a, false, bw,
                                                 (short)0, acc, false, false);
  }
  _Float16* orow = out + (size_t)row0 * CDIM + c0 + n;
  #pragma unroll
  for (int v = 0; v < 8; ++v)
    orow[(size_t)(v + 8 * g) * CDIM] = (_Float16)(acc[v] * scale);
}

// -------- same GEMM but stores transposed: vt[(b*C + c), nsp] ---------------
// Per lane the 8 accumulators are 8 consecutive spatial positions of one
// channel -> a single aligned 16-byte store (global_store_b128).
__global__ __launch_bounds__(128)
void gemm_wt_tr_kernel(const _Float16* __restrict__ A,
                       const _Float16* __restrict__ W,
                       _Float16* __restrict__ vt) {   // [B*C, N]
  __shared__ _Float16 sA[16][CDIM];
  const int row0 = blockIdx.x * 16;
  const int wave = threadIdx.x >> 5;
  const int lane = threadIdx.x & 31;
  for (int i = threadIdx.x; i < 16 * 64; i += 128) {
    int r = i >> 6, ch = i & 63;
    ((uint4*)&sA[r][0])[ch] = ((const uint4*)(A + (size_t)(row0 + r) * CDIM))[ch];
  }
  __syncthreads();
  const int n = lane & 15, g = lane >> 4, m = lane & 15;
  const int c0 = blockIdx.y * 64 + wave * 16;
  const _Float16* wr = W + (size_t)(c0 + n) * CDIM;
  v8f acc = {};
  for (int k0 = 0; k0 < CDIM; k0 += 32) {
    __builtin_prefetch(wr + k0 + 128, 0, 3);
    v16h a, bw;
    #pragma unroll
    for (int v = 0; v < 8; ++v) {
      int ka = g * 16 + 2 * v;
      a[2 * v]     = sA[m][k0 + ka];
      a[2 * v + 1] = sA[m][k0 + ka + 1];
      int kb = g * 16 + 2 * v;
      bw[2 * v]     = wr[k0 + kb];
      bw[2 * v + 1] = wr[k0 + kb + 1];
    }
    acc = __builtin_amdgcn_wmma_f32_16x16x32_f16(false, a, false, bw,
                                                 (short)0, acc, false, false);
  }
  const int c = c0 + n;                          // output channel (= h*64+d)
  const int b = row0 >> 10;                      // 16-row tile stays in one b
  const int nsp0 = (row0 & (NSEQ - 1)) + 8 * g;  // 8 consecutive positions
  alignas(16) _Float16 tmp[8];
  #pragma unroll
  for (int v = 0; v < 8; ++v) tmp[v] = (_Float16)acc[v];
  _Float16* dst = vt + ((size_t)b * CDIM + c) * NSEQ + nsp0;
  *(uint4*)dst = *(const uint4*)tmp;
}

// ---------------- fused attention with dual-top-k softmax -------------------
__global__ __launch_bounds__(128)
void attn_kernel(const _Float16* __restrict__ qh,   // [B,N,C], q pre-scaled d^-1/2
                 const _Float16* __restrict__ kh,   // [B,N,C]
                 const _Float16* __restrict__ vt,   // [B*C, N] (pre-transposed V)
                 _Float16* __restrict__ ao,         // [B,N,C]
                 const int* __restrict__ kvals) {
  __shared__ float sS[16][NSEQ + 1];                 // score rows, padded
  __shared__ unsigned hist[4][256];                  // per-wave radix hist
  const int bh = blockIdx.x;
  const int b = bh / NHEADS, h = bh % NHEADS;
  const int row0 = blockIdx.y * 16;
  const int wave = threadIdx.x >> 5;
  const int lane = threadIdx.x & 31;
  const int k1 = kvals[0], k2 = kvals[1];

  const _Float16* qbase = qh + (size_t)b * NSEQ * CDIM + h * HD;
  const _Float16* kbase = kh + (size_t)b * NSEQ * CDIM + h * HD;
  const _Float16* vtbase = vt + ((size_t)b * CDIM + h * HD) * NSEQ;

  // Q A-fragments: 16x64 = two 16x32 frags (replicated per wave)
  v16h qa0, qa1;
  {
    int m = lane & 15, g = lane >> 4;
    const _Float16* qr = qbase + (size_t)(row0 + m) * CDIM;
    #pragma unroll
    for (int v = 0; v < 8; ++v) {
      int ka = g * 16 + 2 * v;
      qa0[2 * v] = qr[ka];      qa0[2 * v + 1] = qr[ka + 1];
      qa1[2 * v] = qr[32 + ka]; qa1[2 * v + 1] = qr[32 + ka + 1];
    }
  }

  // S = Q @ K^T : each wave owns every 4th 16-key tile
  {
    int n = lane & 15, g = lane >> 4;
    for (int jt = wave; jt < NSEQ / 16; jt += 4) {
      const int n0 = jt * 16;
      const _Float16* kr = kbase + (size_t)(n0 + n) * CDIM;  // B(k,n)=K[n0+n][k]
      __builtin_prefetch(kbase + (size_t)(n0 + 64 + n) * CDIM, 0, 3);
      v16h kb0, kb1;
      #pragma unroll
      for (int v = 0; v < 8; ++v) {
        int kk = g * 16 + 2 * v;
        kb0[2 * v] = kr[kk];      kb0[2 * v + 1] = kr[kk + 1];
        kb1[2 * v] = kr[32 + kk]; kb1[2 * v + 1] = kr[32 + kk + 1];
      }
      v8f acc = {};
      acc = __builtin_amdgcn_wmma_f32_16x16x32_f16(false, qa0, false, kb0,
                                                   (short)0, acc, false, false);
      acc = __builtin_amdgcn_wmma_f32_16x16x32_f16(false, qa1, false, kb1,
                                                   (short)0, acc, false, false);
      #pragma unroll
      for (int v = 0; v < 8; ++v) sS[v + 8 * g][n0 + n] = acc[v];
    }
  }
  __syncthreads();

  // per-row: max, dual radix-select thresholds, combined masked-softmax weights
  for (int rr = 0; rr < 4; ++rr) {
    const int row = wave * 4 + rr;
    float* rowp = &sS[row][0];
    float mx = -3.4e38f;
    for (int j = lane; j < NSEQ; j += 32) mx = fmaxf(mx, rowp[j]);
    mx = wave_max(mx);
    float thr1 = radix_kth_largest(rowp, NSEQ, k1, &hist[wave][0], lane);
    float thr2 = radix_kth_largest(rowp, NSEQ, k2, &hist[wave][0], lane);
    float z1 = 0.0f, z2 = 0.0f;
    for (int j = lane; j < NSEQ; j += 32) {
      float s = rowp[j];
      float e = __expf(s - mx);
      if (s >= thr1) z1 += e;
      if (s >= thr2) z2 += e;
    }
    z1 = wave_sum(z1); z2 = wave_sum(z2);
    const float c1 = 0.6f / z1, c2 = 0.4f / z2;
    for (int j = lane; j < NSEQ; j += 32) {
      float s = rowp[j];
      float e = __expf(s - mx);
      float w = 0.0f;
      if (s >= thr1) w += c1 * e;
      if (s >= thr2) w += c2 * e;
      rowp[j] = w;
    }
  }
  __syncthreads();

  // O = P @ V : wave owns 16 output columns, 32 chained WMMAs over K=1024.
  // V^T rows are contiguous in the key index -> b128 fragment loads.
  {
    int n = lane & 15, g = lane >> 4, m = lane & 15;
    const _Float16* vr0 = vtbase + (size_t)(wave * 16 + n) * NSEQ;
    v8f acc = {};
    for (int kt = 0; kt < NSEQ / 32; ++kt) {
      v16h pa, vb;
      #pragma unroll
      for (int v = 0; v < 8; ++v) {
        int ka = g * 16 + 2 * v;
        pa[2 * v]     = (_Float16)sS[m][kt * 32 + ka];
        pa[2 * v + 1] = (_Float16)sS[m][kt * 32 + ka + 1];
      }
      const _Float16* vr = vr0 + kt * 32 + g * 16;   // 16 consecutive halves
      #pragma unroll
      for (int v = 0; v < 8; ++v) {
        vb[2 * v]     = vr[2 * v];
        vb[2 * v + 1] = vr[2 * v + 1];
      }
      acc = __builtin_amdgcn_wmma_f32_16x16x32_f16(false, pa, false, vb,
                                                   (short)0, acc, false, false);
    }
    _Float16* aor = ao + ((size_t)b * NSEQ + row0) * CDIM + h * HD + wave * 16 + n;
    #pragma unroll
    for (int v = 0; v < 8; ++v)
      aor[(size_t)(v + 8 * g) * CDIM] = (_Float16)acc[v];
  }
}

// ---------------- final projection + bias + residual + NCHW transpose -------
__global__ __launch_bounds__(128)
void proj_kernel(const _Float16* __restrict__ A,     // ao [8192,512]
                 const _Float16* __restrict__ W,     // proj_w f16 [512,512]
                 const float* __restrict__ bias,
                 const float* __restrict__ x,        // residual, [B,C,N]
                 float* __restrict__ outp) {         // [B,C,N]
  __shared__ _Float16 sA[16][CDIM];
  const int row0 = blockIdx.x * 16;
  const int wave = threadIdx.x >> 5;
  const int lane = threadIdx.x & 31;
  for (int i = threadIdx.x; i < 16 * 64; i += 128) {
    int r = i >> 6, ch = i & 63;
    ((uint4*)&sA[r][0])[ch] = ((const uint4*)(A + (size_t)(row0 + r) * CDIM))[ch];
  }
  __syncthreads();
  const int n = lane & 15, g = lane >> 4, m = lane & 15;
  const int c0 = blockIdx.y * 64 + wave * 16;
  const _Float16* wr = W + (size_t)(c0 + n) * CDIM;
  v8f acc = {};
  for (int k0 = 0; k0 < CDIM; k0 += 32) {
    __builtin_prefetch(wr + k0 + 128, 0, 3);
    v16h a, bw;
    #pragma unroll
    for (int v = 0; v < 8; ++v) {
      int ka = g * 16 + 2 * v;
      a[2 * v]     = sA[m][k0 + ka];
      a[2 * v + 1] = sA[m][k0 + ka + 1];
      int kb = g * 16 + 2 * v;
      bw[2 * v]     = wr[k0 + kb];
      bw[2 * v + 1] = wr[k0 + kb + 1];
    }
    acc = __builtin_amdgcn_wmma_f32_16x16x32_f16(false, a, false, bw,
                                                 (short)0, acc, false, false);
  }
  const int c = c0 + n;
  const float bi = bias[c];
  #pragma unroll
  for (int v = 0; v < 8; ++v) {
    int row = row0 + v + 8 * g;           // row = b*1024 + n_spatial
    int b = row >> 10, nsp = row & (NSEQ - 1);
    size_t oi = ((size_t)b * CDIM + c) * NSEQ + nsp;
    outp[oi] = acc[v] + bi + x[oi];
  }
}

// ---------------------------------------------------------------------------
extern "C" void kernel_launch(void* const* d_in, const int* in_sizes, int n_in,
                              void* d_out, int out_size, void* d_ws, size_t ws_size,
                              hipStream_t stream) {
  (void)in_sizes; (void)n_in; (void)out_size; (void)ws_size;
  const float* x      = (const float*)d_in[0];
  const float* y      = (const float*)d_in[1];
  const float* q_w    = (const float*)d_in[2];
  const float* kv_w   = (const float*)d_in[3];
  const float* proj_w = (const float*)d_in[4];
  const float* proj_b = (const float*)d_in[5];
  const float* c1w = (const float*)d_in[6];
  const float* c1b = (const float*)d_in[7];
  const float* c2w = (const float*)d_in[8];
  const float* c2b = (const float*)d_in[9];
  const float* c3w = (const float*)d_in[10];
  const float* c3b = (const float*)d_in[11];
  const float* lnw = (const float*)d_in[12];
  const float* lnb = (const float*)d_in[13];
  const float* kr1 = (const float*)d_in[14];
  const float* kr2 = (const float*)d_in[15];
  float* out = (float*)d_out;

  char* base = (char*)d_ws;
  size_t off = 0;
  auto alloc = [&](size_t bytes) {
    char* p = base + off;
    off += (bytes + 255) & ~(size_t)255;
    return p;
  };
  const size_t BNC = (size_t)BDIM * NSEQ * CDIM;
  _Float16* qwh  = (_Float16*)alloc((size_t)CDIM * CDIM * 2);
  _Float16* kvwh = (_Float16*)alloc((size_t)2 * CDIM * CDIM * 2);
  _Float16* pwh  = (_Float16*)alloc((size_t)CDIM * CDIM * 2);
  _Float16* xsh  = (_Float16*)alloc(BNC * 2);
  _Float16* ynh  = (_Float16*)alloc(BNC * 2);
  _Float16* qhp  = (_Float16*)alloc(BNC * 2);
  _Float16* khp  = (_Float16*)alloc(BNC * 2);
  _Float16* vth  = (_Float16*)alloc(BNC * 2);   // V pre-transposed [B*C, N]
  _Float16* aoh  = (_Float16*)alloc(BNC * 2);
  int* kvals     = (int*)alloc(16);

  cvt_w_kernel<<<2048, 256, 0, stream>>>(q_w, kv_w, proj_w, qwh, kvwh, pwh);
  compute_k_kernel<<<1, 32, 0, stream>>>(kr1, kr2, kvals);
  transpose_x_kernel<<<16384, 256, 0, stream>>>(x, xsh);
  conv_ln_kernel<<<BDIM * NSEQ, CDIM, 0, stream>>>(y, c1w, c1b, c2w, c2b,
                                                   c3w, c3b, lnw, lnb, ynh);
  // q pre-scaled by hd^-0.5 = 0.125
  gemm_wt_kernel<<<dim3(MROWS / 16, 8), 128, 0, stream>>>(xsh, qwh, qhp, 0.125f);
  gemm_wt_kernel<<<dim3(MROWS / 16, 8), 128, 0, stream>>>(ynh, kvwh, khp, 1.0f);
  gemm_wt_tr_kernel<<<dim3(MROWS / 16, 8), 128, 0, stream>>>(
      ynh, kvwh + (size_t)CDIM * CDIM, vth);
  attn_kernel<<<dim3(BDIM * NHEADS, NSEQ / 16), 128, 0, stream>>>(
      qhp, khp, vth, aoh, kvals);
  proj_kernel<<<dim3(MROWS / 16, 8), 128, 0, stream>>>(aoh, pwh, proj_b, x, out);
}
